// Inception1d_62680752718395
// MI455X (gfx1250) — compile-verified, matrix operand order checked by reading, not run on hardware
//
#include <hip/hip_runtime.h>
#include <hip/hip_bf16.h>

typedef __attribute__((ext_vector_type(16))) __bf16 v16bf;
typedef __attribute__((ext_vector_type(8)))  float  v8f;

#define L_LEN   16384
#define C_IN    64
#define N_OC    36          // real output channels
#define KSTEPS  30          // 15 taps * 64 ci / 32
#define APACK_ELEMS (3 * KSTEPS * 32 * 16)   // 46080 bf16 = 92160 bytes

static __device__ __forceinline__ unsigned short f2bf(float f) {
    unsigned int u = __float_as_uint(f);
    unsigned int r = (u + 0x7FFFu + ((u >> 16) & 1u)) >> 16;  // RNE
    return (unsigned short)r;
}

// ---------------------------------------------------------------------------
// Kernel 1: fuse the 8 branch weights into a single [48 x (64ci x 15tap)]
// bf16 matrix, pre-swizzled into the exact 16-bit A-matrix VGPR layout
// (lanes 0-15: K 0-7 / 16-23, lanes 16-31: K 8-15 / 24-31 per 32-K step).
// K ordering: k = tap*64 + ci. Also packs the fused bias[36].
// ---------------------------------------------------------------------------
__global__ void pack_weights_kernel(
    const float* __restrict__ w1,  const float* __restrict__ w3,
    const float* __restrict__ w5,  const float* __restrict__ w7,
    const float* __restrict__ w9,  const float* __restrict__ w11,
    const float* __restrict__ w13, const float* __restrict__ w15,
    const float* __restrict__ b1,  const float* __restrict__ b3,
    const float* __restrict__ b5,  const float* __restrict__ b7,
    const float* __restrict__ b9,  const float* __restrict__ b11,
    const float* __restrict__ b13, const float* __restrict__ b15,
    unsigned short* __restrict__ Apack, float* __restrict__ biasOut)
{
    const float* wArr[8] = {w1, w3, w5, w7, w9, w11, w13, w15};
    const float* bArr[8] = {b1, b3, b5, b7, b9, b11, b13, b15};

    int e = blockIdx.x * blockDim.x + threadIdx.x;
    if (e < APACK_ELEMS) {
        int i    = e & 15;          // half index inside fragment
        int lane = (e >> 4) & 31;
        int ks   = (e >> 9) % KSTEPS;
        int mt   = e / (KSTEPS * 32 * 16);
        int m  = mt * 16 + (lane & 15);
        int hi = lane >> 4;
        int k_local = hi * 8 + (i < 8 ? i : i + 8);  // 0-7/16-23 or 8-15/24-31
        int t  = ks >> 1;                     // tap 0..14
        int ci = (ks & 1) * 32 + k_local;     // channel 0..63
        float val = 0.f;
        if (m < N_OC) {
            int bi, oc;
            if (m == 0) { bi = 0; oc = 0; }
            else        { bi = 1 + (m - 1) / 5; oc = (m - 1) % 5; }
            int kk = 2 * bi + 1;                  // kernel size 1,3,...,15
            int j  = t - 7 + (kk - 1) / 2;        // tap index inside branch
            if (j >= 0 && j < kk)
                val = wArr[bi][(oc * C_IN + ci) * kk + j];
        }
        Apack[e] = f2bf(val);
    }
    if (blockIdx.x == 0 && threadIdx.x < N_OC) {
        int m = threadIdx.x;
        int bi, oc;
        if (m == 0) { bi = 0; oc = 0; }
        else        { bi = 1 + (m - 1) / 5; oc = (m - 1) % 5; }
        biasOut[m] = bArr[bi][oc];
    }
}

// ---------------------------------------------------------------------------
// Kernel 2: fused inception conv as implicit GEMM with v_wmma_f32_16x16x32_bf16.
// One workgroup = 1 batch row x 256 output positions. x tile (270 pos x 64 ci,
// halo +-7, zero-padded at row edges) staged in LDS as bf16, transposed to
// [pos][ci] with 144B row stride so B fragments are aligned b128 pairs.
// 8 waves x (3 M-tiles x 2 N-subtiles) accumulators; 30 K-steps of 32.
// ---------------------------------------------------------------------------
__global__ __launch_bounds__(256) void inception_wmma_kernel(
    const float* __restrict__ x,
    const unsigned short* __restrict__ Apack,
    const float* __restrict__ bias,
    float* __restrict__ out)
{
    __shared__ __align__(16) unsigned short lds[270 * 72];  // 38.9 KB

    const int tid = threadIdx.x;
    const int b   = blockIdx.y;
    const int l0  = blockIdx.x * 256;

    // Stage x -> LDS (fp32 load, bf16 store, transpose to [pos][ci]).
    for (int idx = tid; idx < 270 * C_IN; idx += 256) {
        int ci = idx / 270;
        int p  = idx - ci * 270;
        int gpos = l0 - 7 + p;
        float v = 0.f;
        if (gpos >= 0 && gpos < L_LEN)
            v = x[((size_t)b * C_IN + ci) * L_LEN + gpos];
        lds[p * 72 + ci] = f2bf(v);
    }
    __syncthreads();

    const int wave = tid >> 5;
    const int lane = tid & 31;
    const int n    = lane & 15;
    const int hi   = lane >> 4;
    const int ciB  = hi * 8;   // within-32 K chunk base (A/B 16-bit layout)

    v8f acc[3][2] = {};

    for (int ks = 0; ks < KSTEPS; ++ks) {
        int t    = ks >> 1;
        int cblk = (ks & 1) * 32;

        v16bf a[3];
#pragma unroll
        for (int mt = 0; mt < 3; ++mt) {
            const uint4* ap =
                (const uint4*)(Apack + ((size_t)(mt * KSTEPS + ks) * 32 + lane) * 16);
            ((uint4*)&a[mt])[0] = ap[0];
            ((uint4*)&a[mt])[1] = ap[1];
        }
#pragma unroll
        for (int j = 0; j < 2; ++j) {
            int outn = (2 * wave + j) * 16 + n;   // output position in tile
            int pos  = outn + t;                  // staged-tile input row
            const unsigned short* bp = &lds[pos * 72 + cblk + ciB];
            v16bf bf;
            ((uint4*)&bf)[0] = *(const uint4*)(bp);
            ((uint4*)&bf)[1] = *(const uint4*)(bp + 16);
#pragma unroll
            for (int mt = 0; mt < 3; ++mt)
                acc[mt][j] = __builtin_amdgcn_wmma_f32_16x16x32_bf16(
                    false, a[mt], false, bf, (short)0, acc[mt][j], false, false);
        }
    }

    // Epilogue: bias + store 36 real channels (C/D layout: M = r + 8*hi).
#pragma unroll
    for (int mt = 0; mt < 3; ++mt) {
#pragma unroll
        for (int j = 0; j < 2; ++j) {
            int outn = l0 + (2 * wave + j) * 16 + n;
#pragma unroll
            for (int r = 0; r < 8; ++r) {
                int m = mt * 16 + hi * 8 + r;
                if (m < N_OC)
                    out[((size_t)b * N_OC + m) * L_LEN + outn] = acc[mt][j][r] + bias[m];
            }
        }
    }
}

// ---------------------------------------------------------------------------
extern "C" void kernel_launch(void* const* d_in, const int* in_sizes, int n_in,
                              void* d_out, int out_size, void* d_ws, size_t ws_size,
                              hipStream_t stream) {
    (void)in_sizes; (void)n_in; (void)out_size; (void)ws_size;
    const float* x = (const float*)d_in[0];
    const float* w[8];
    const float* bb[8];
    for (int i = 0; i < 8; ++i) {
        w[i]  = (const float*)d_in[1 + 2 * i];
        bb[i] = (const float*)d_in[2 + 2 * i];
    }

    unsigned short* Apack = (unsigned short*)d_ws;
    float* bias = (float*)((char*)d_ws + (size_t)APACK_ELEMS * sizeof(unsigned short));

    pack_weights_kernel<<<(APACK_ELEMS + 255) / 256, 256, 0, stream>>>(
        w[0], w[1], w[2], w[3], w[4], w[5], w[6], w[7],
        bb[0], bb[1], bb[2], bb[3], bb[4], bb[5], bb[6], bb[7],
        Apack, bias);

    dim3 grid(L_LEN / 256, 64);
    inception_wmma_kernel<<<grid, 256, 0, stream>>>(x, Apack, bias, (float*)d_out);
}
